// HeteroTransformerEncoderLayer_13443247637014
// MI455X (gfx1250) — compile-verified, matrix operand order checked by reading, not run on hardware
//
#include <hip/hip_runtime.h>

// ---------------------------------------------------------------------------
// MI455X (gfx1250) per-token transformer encoder layer.
// Matmuls: v_wmma_f32_16x16x32_bf16 (wave32 matrix cores), f32 accumulate.
// A-operand tiles staged to LDS via GLOBAL_LOAD_ASYNC_TO_LDS_B128 (ASYNCcnt);
// W tiles staged f32->bf16 with vectorized float4 loads.
// ---------------------------------------------------------------------------

typedef __attribute__((ext_vector_type(16))) __bf16 v16bf;
typedef __attribute__((ext_vector_type(8)))  __bf16 v8bf;
typedef __attribute__((ext_vector_type(8)))  float  v8f;
typedef int v4i __attribute__((vector_size(16)));

constexpr int Bc   = 256;   // batch
constexpr int Tc   = 128;   // tokens
constexpr int Dc   = 256;   // model dim
constexpr int Hc   = 8;     // heads
constexpr int AEc  = 32;    // attn head dim
constexpr int DFFc = 512;   // ffn hidden
constexpr float EPSc = 1e-5f;

static __device__ __forceinline__ v8f wmma_bf16(v16bf a, v16bf b, v8f c) {
  return __builtin_amdgcn_wmma_f32_16x16x32_bf16(false, a, false, b,
                                                 (short)0, c, false, false);
}

// 16-byte global->LDS copy; async DMA path (ASYNCcnt) when available.
static __device__ __forceinline__ void copy_b128_to_lds(const __bf16* g,
                                                        __bf16* l) {
#if __has_builtin(__builtin_amdgcn_global_load_async_to_lds_b128)
  __builtin_amdgcn_global_load_async_to_lds_b128(
      (__attribute__((address_space(1))) v4i*)g,
      (__attribute__((address_space(3))) v4i*)l, 0, 0);
#else
  *(v8bf*)l = *(const v8bf*)g;
#endif
}

static __device__ __forceinline__ void copy_wait() {
#if __has_builtin(__builtin_amdgcn_global_load_async_to_lds_b128)
#if __has_builtin(__builtin_amdgcn_s_wait_asynccnt)
  __builtin_amdgcn_s_wait_asynccnt(0);
#else
  asm volatile("s_wait_asynccnt 0x0" ::: "memory");
#endif
#endif
}

// Load a 16x32 fragment (rows r0..r0+15, k0..k0+31) from a row-major bf16
// buffer with row stride `ld` (elements). CDNA5 16-bit A layout:
// lane L holds row r0+(L&15); elems 0..7 = K (L>>4)*8+0..7,
// elems 8..15 = K 16+(L>>4)*8+0..7. Two 16-byte loads per lane.
static __device__ __forceinline__ v16bf ld_frag(const __bf16* base, int ld,
                                                int r0, int k0) {
  const int lane = threadIdx.x & 31;
  const __bf16* p =
      base + (size_t)(r0 + (lane & 15)) * (size_t)ld + k0 + ((lane >> 4) << 3);
  union { v16bf v; v8bf h[2]; } u;
  u.h[0] = *(const v8bf*)(p);
  u.h[1] = *(const v8bf*)(p + 16);
  return u.v;
}

// ---------------------------------------------------------------------------
// Generic per-token GEMM:  C[b, t, n] = sum_k A[b, t, k] * W[t, k, n]
// A: bf16 (B, T, Ka).  W: f32 (T, Ka, N).
// Block = 256 threads (8 waves). Macro tile 128(M=batch) x 64(N).
// Wave (wy = w>>1, wx = w&1) owns rows [wy*32, wy*32+32) x cols [wx*32, +32)
// -> 4 WMMAs per k-step.
// mode 0: store bf16 natural (B,T,N)
// mode 1: store bf16 V-transposed: [((b*H + n>>8)*256 + (n&255))*T + t]
// mode 2: store f32 natural with residual add (Rf, same shape)
// ---------------------------------------------------------------------------
__global__ __launch_bounds__(256) void tok_gemm(
    const __bf16* __restrict__ A, int Ka,
    const float* __restrict__ W, int N,
    __bf16* __restrict__ Cb, float* __restrict__ Cf,
    const float* __restrict__ Rf, int mode) {
  const int t   = blockIdx.z;
  const int n0  = blockIdx.x * 64;
  const int m0  = blockIdx.y * 128;
  const int tid = threadIdx.x;
  const int w = tid >> 5, lane = tid & 31;
  const int wy = w >> 1, wx = w & 1;

  __shared__ __align__(16) __bf16 sA[128][40];  // [m][k] tile, padded
  __shared__ __align__(16) __bf16 sW[64][40];   // [n][k] transposed, padded

  v8f acc00 = {}, acc01 = {}, acc10 = {}, acc11 = {};
  const __bf16* Abase = A + (size_t)t * Ka;  // row m -> + m*(T*Ka)
  const int Astride = Tc * Ka;

  for (int k0 = 0; k0 < Ka; k0 += 32) {
    // --- stage A tile (128 rows x 32 k, bf16): 512 x 16B async copies ---
#pragma unroll
    for (int r = 0; r < 2; ++r) {
      const int task = tid + 256 * r;
      const int row = task >> 2;          // 0..127
      const int seg = (task & 3) << 3;    // element offset 0,8,16,24
      copy_b128_to_lds(Abase + (size_t)(m0 + row) * Astride + k0 + seg,
                       &sA[row][seg]);
    }
    // --- stage W tile (32 k x 64 n, f32 -> bf16 transposed) ---
    {
      const int nn = (tid & 15) << 2;     // 0..60 step 4
      for (int kk = tid >> 4; kk < 32; kk += 16) {
        const float4 wv =
            *(const float4*)&W[((size_t)t * Ka + (k0 + kk)) * (size_t)N +
                               (n0 + nn)];
        sW[nn + 0][kk] = (__bf16)wv.x;
        sW[nn + 1][kk] = (__bf16)wv.y;
        sW[nn + 2][kk] = (__bf16)wv.z;
        sW[nn + 3][kk] = (__bf16)wv.w;
      }
    }
    copy_wait();
    __syncthreads();

    v16bf a0 = ld_frag(&sA[0][0], 40, wy * 32, 0);
    v16bf a1 = ld_frag(&sA[0][0], 40, wy * 32 + 16, 0);
    v16bf b0 = ld_frag(&sW[0][0], 40, wx * 32, 0);
    v16bf b1 = ld_frag(&sW[0][0], 40, wx * 32 + 16, 0);
    acc00 = wmma_bf16(a0, b0, acc00);
    acc01 = wmma_bf16(a0, b1, acc01);
    acc10 = wmma_bf16(a1, b0, acc10);
    acc11 = wmma_bf16(a1, b1, acc11);
    __syncthreads();
  }

  // Epilogue. C layout: VGPR j -> row (lane>>4)*8 + j, col lane&15.
  const int cbase = n0 + wx * 32 + (lane & 15);
#pragma unroll
  for (int ai = 0; ai < 2; ++ai) {
    const int mbase = m0 + wy * 32 + ai * 16 + ((lane >> 4) << 3);
#pragma unroll
    for (int half = 0; half < 2; ++half) {
      const int col = cbase + half * 16;
#pragma unroll
      for (int j = 0; j < 8; ++j) {
        const int m = mbase + j;
        const float val = ai ? (half ? acc11[j] : acc10[j])
                             : (half ? acc01[j] : acc00[j]);
        if (mode == 0) {
          Cb[((size_t)m * Tc + t) * (size_t)N + col] = (__bf16)val;
        } else if (mode == 1) {
          Cb[((size_t)(m * Hc + (col >> 8)) * 256 + (col & 255)) * Tc + t] =
              (__bf16)val;
        } else {
          const size_t idx = ((size_t)m * Tc + t) * (size_t)N + col;
          Cf[idx] = val + Rf[idx];
        }
      }
    }
  }
}

// ---------------------------------------------------------------------------
// Attention: one block per (b, h). Wave w owns token rows [w*16, w*16+16).
// scores = q k^T / sqrt(AE) via 1 WMMA per 16x16 tile (K=AE=32),
// softmax in LDS, then out = P @ V (K=128) with V pre-transposed in global.
// ---------------------------------------------------------------------------
__global__ __launch_bounds__(256) void attn_kernel(
    const __bf16* __restrict__ qb, const __bf16* __restrict__ kb,
    const __bf16* __restrict__ vT, __bf16* __restrict__ ob) {
  const int bh = blockIdx.x;
  const int b = bh >> 3, h = bh & 7;
  const int w = threadIdx.x >> 5, lane = threadIdx.x & 31;
  const int t0 = w * 16;

  __shared__ __align__(16) float  sS[8][16][132];
  __shared__ __align__(16) __bf16 sP[8][16][136];

  const __bf16* qbase = qb + ((size_t)b * Tc) * 256 + h * AEc;  // ld 256
  const __bf16* kbase = kb + ((size_t)b * Tc) * 256 + h * AEc;  // ld 256

  // --- scores + scale ---
  const float scale = 0.17677669529663687f;  // 1/sqrt(32)
#pragma unroll
  for (int nt = 0; nt < 8; ++nt) {
    v8f acc = {};
    v16bf aq = ld_frag(qbase, 256, t0, 0);
    v16bf bk = ld_frag(kbase, 256, nt * 16, 0);
    acc = wmma_bf16(aq, bk, acc);
    const int rl = (lane >> 4) << 3;
    const int cl = nt * 16 + (lane & 15);
#pragma unroll
    for (int j = 0; j < 8; ++j) sS[w][rl + j][cl] = acc[j] * scale;
  }
  __syncthreads();

  // --- softmax: lanes 0..15 each own one row of the slab ---
  if (lane < 16) {
    float mx = -1e30f;
    for (int c = 0; c < Tc; ++c) mx = fmaxf(mx, sS[w][lane][c]);
    float sum = 0.f;
    for (int c = 0; c < Tc; ++c) {
      const float e = __expf(sS[w][lane][c] - mx);
      sS[w][lane][c] = e;
      sum += e;
    }
    const float inv = 1.f / sum;
    for (int c = 0; c < Tc; ++c)
      sP[w][lane][c] = (__bf16)(sS[w][lane][c] * inv);
  }
  __syncthreads();

  // --- out = P (16x128) @ V (128x256), V fragments direct from global ---
  const __bf16* vbase = vT + ((size_t)(b * Hc + h)) * 256 * Tc;  // ld 128
  __bf16* obase = ob + ((size_t)b * Tc) * 2048 + h * 256;
#pragma unroll 4
  for (int nt = 0; nt < 16; ++nt) {
    v8f acc = {};
#pragma unroll
    for (int k0 = 0; k0 < Tc; k0 += 32) {
      v16bf ap = ld_frag(&sP[w][0][0], 136, 0, k0);
      v16bf bv = ld_frag(vbase, Tc, nt * 16, k0);
      acc = wmma_bf16(ap, bv, acc);
    }
    const int rl = t0 + ((lane >> 4) << 3);
    const int cl = nt * 16 + (lane & 15);
#pragma unroll
    for (int j = 0; j < 8; ++j)
      obase[(size_t)(rl + j) * 2048 + cl] = (__bf16)acc[j];
  }
}

// --------------------------- pointwise kernels -----------------------------
__global__ __launch_bounds__(256) void k_cvt_bf16(const float* __restrict__ x,
                                                  __bf16* __restrict__ y,
                                                  int n) {
  const int i = blockIdx.x * 256 + threadIdx.x;
  if (i < n) y[i] = (__bf16)x[i];
}

__global__ __launch_bounds__(256) void k_silu_mul(const __bf16* __restrict__ g,
                                                  const __bf16* __restrict__ u,
                                                  __bf16* __restrict__ o,
                                                  int n) {
  const int i = blockIdx.x * 256 + threadIdx.x;
  if (i < n) {
    const float gf = (float)g[i];
    const float uf = (float)u[i];
    o[i] = (__bf16)((gf / (1.f + __expf(-gf))) * uf);
  }
}

// LayerNorm over rows of length 256. One block per row, 256 threads.
__global__ __launch_bounds__(256) void k_layernorm(
    const float* __restrict__ Y, const float* __restrict__ gamma,
    const float* __restrict__ beta, float* __restrict__ Of,
    __bf16* __restrict__ Ob) {
  __shared__ float red[256];
  const int row = blockIdx.x, d = threadIdx.x;
  const float v = Y[(size_t)row * Dc + d];
  red[d] = v;
  __syncthreads();
  for (int s = 128; s > 0; s >>= 1) {
    if (d < s) red[d] += red[d + s];
    __syncthreads();
  }
  const float mu = red[0] * (1.f / Dc);
  __syncthreads();
  const float dv = v - mu;
  red[d] = dv * dv;
  __syncthreads();
  for (int s = 128; s > 0; s >>= 1) {
    if (d < s) red[d] += red[d + s];
    __syncthreads();
  }
  const float var = red[0] * (1.f / Dc);
  const float out = dv * rsqrtf(var + EPSc) * gamma[d] + beta[d];
  Of[(size_t)row * Dc + d] = out;
  if (Ob) Ob[(size_t)row * Dc + d] = (__bf16)out;
}

// ---------------------------------------------------------------------------
extern "C" void kernel_launch(void* const* d_in, const int* in_sizes, int n_in,
                              void* d_out, int out_size, void* d_ws,
                              size_t ws_size, hipStream_t stream) {
  (void)in_sizes; (void)n_in; (void)out_size; (void)ws_size;
  const float* X  = (const float*)d_in[0];
  const float* Q  = (const float*)d_in[1];
  const float* K  = (const float*)d_in[2];
  const float* V  = (const float*)d_in[3];
  const float* O  = (const float*)d_in[4];
  const float* Wg = (const float*)d_in[5];
  const float* Wu = (const float*)d_in[6];
  const float* Wd = (const float*)d_in[7];
  const float* g1 = (const float*)d_in[8];
  const float* b1 = (const float*)d_in[9];
  const float* g2 = (const float*)d_in[10];
  const float* b2 = (const float*)d_in[11];
  float* OUT = (float*)d_out;

  // workspace carve-out (256B aligned), with dead-buffer overlays
  char* wsp = (char*)d_ws;
  size_t off = 0;
  auto carve = [&](size_t bytes) {
    void* p = wsp + off;
    off = (off + bytes + 255) & ~(size_t)255;
    return p;
  };
  const size_t szBTD2 = (size_t)Bc * Tc * Dc * 2;    // 16.78 MB
  const size_t szBTF2 = (size_t)Bc * Tc * DFFc * 2;  // 33.55 MB
  const size_t szBTN2 = (size_t)Bc * Tc * 2048 * 2;  // 134.2 MB
  const size_t szBTD4 = (size_t)Bc * Tc * Dc * 4;    // 33.55 MB

  __bf16* Xb  = (__bf16*)carve(szBTD2);
  __bf16* qb  = (__bf16*)carve(szBTD2);
  __bf16* kb  = (__bf16*)carve(szBTD2);
  char*   vTr = (char*)carve(szBTN2);  // vT; reused for gb/ub/hb after attn
  __bf16* vT  = (__bf16*)vTr;
  __bf16* ob  = (__bf16*)carve(szBTN2);
  float*  y   = (float*)carve(szBTD4);  // reused as z after LN1
  float*  x   = (float*)carve(szBTD4);
  __bf16* xb  = (__bf16*)carve(szBTD2);
  __bf16* gb  = (__bf16*)(vTr);
  __bf16* ub  = (__bf16*)(vTr + szBTF2);
  __bf16* hb  = (__bf16*)(vTr + 2 * szBTF2);
  float*  z   = y;

  // 1) X -> bf16
  k_cvt_bf16<<<(Bc * Tc * Dc) / 256, 256, 0, stream>>>(X, Xb, Bc * Tc * Dc);
  // 2) q = X @ Q[t], k = X @ K[t]   (N=256)
  tok_gemm<<<dim3(4, 2, Tc), 256, 0, stream>>>(Xb, Dc, Q, 256, qb, nullptr,
                                               nullptr, 0);
  tok_gemm<<<dim3(4, 2, Tc), 256, 0, stream>>>(Xb, Dc, K, 256, kb, nullptr,
                                               nullptr, 0);
  // 3) v = X @ V[t]  (N=2048), stored transposed per (b,h): [dd][t]
  tok_gemm<<<dim3(32, 2, Tc), 256, 0, stream>>>(Xb, Dc, V, 2048, vT, nullptr,
                                                nullptr, 1);
  // 4) attention
  attn_kernel<<<Bc * Hc, 256, 0, stream>>>(qb, kb, vT, ob);
  // 5) y = X + out @ O[t]   (Ka=2048, N=256, f32 + residual)
  tok_gemm<<<dim3(4, 2, Tc), 256, 0, stream>>>(ob, 2048, O, 256, nullptr, y, X,
                                               2);
  // 6) x = LN1(y); also bf16 copy
  k_layernorm<<<Bc * Tc, 256, 0, stream>>>(y, g1, b1, x, xb);
  // 7) gate / up  (N=512)
  tok_gemm<<<dim3(8, 2, Tc), 256, 0, stream>>>(xb, Dc, Wg, DFFc, gb, nullptr,
                                               nullptr, 0);
  tok_gemm<<<dim3(8, 2, Tc), 256, 0, stream>>>(xb, Dc, Wu, DFFc, ub, nullptr,
                                               nullptr, 0);
  // 8) hidden = silu(gate) * up
  k_silu_mul<<<(Bc * Tc * DFFc) / 256, 256, 0, stream>>>(gb, ub, hb,
                                                         Bc * Tc * DFFc);
  // 9) z = x + hidden @ W_down[t]   (Ka=512, N=256)
  tok_gemm<<<dim3(4, 2, Tc), 256, 0, stream>>>(hb, DFFc, Wd, 256, nullptr, z, x,
                                               2);
  // 10) out = LN2(z)
  k_layernorm<<<Bc * Tc, 256, 0, stream>>>(z, g2, b2, OUT, nullptr);
}
